// LinearAttention_53352083751002
// MI455X (gfx1250) — matrix operand order repaired and checked by reading, not compiled, not run on hardware
//
#include <hip/hip_runtime.h>
#include <hip/hip_bf16.h>

typedef __attribute__((ext_vector_type(16))) __bf16 v16bf;
typedef __attribute__((ext_vector_type(8)))  __bf16 v8bf;
typedef __attribute__((ext_vector_type(4)))  __bf16 v4bf;
typedef __attribute__((ext_vector_type(8)))  float  v8f;
typedef __attribute__((ext_vector_type(4)))  float  v4f;
typedef __attribute__((ext_vector_type(4)))  unsigned int u32x4;
typedef __attribute__((ext_vector_type(8)))  int i32x8;
typedef __attribute__((ext_vector_type(4)))  int i32x4;

#define HID   1024
#define NH    16
#define HD    64
#define BATCH 2
#define SEQ   2048
#define MTOT  (BATCH * SEQ)   // 4096

#if __has_builtin(__builtin_amdgcn_tensor_load_to_lds) && __has_builtin(__builtin_amdgcn_s_wait_tensorcnt)
#define HAVE_TDM 1
#else
#define HAVE_TDM 0
#endif

__device__ __forceinline__ v16bf cat8(v8bf lo, v8bf hi) {
  return __builtin_shufflevector(lo, hi, 0,1,2,3,4,5,6,7,8,9,10,11,12,13,14,15);
}

// ---------------- WMMA fragment helpers (wave32 layouts, all b128-vectorized) ---------
__device__ __forceinline__ v16bf load_Afrag(const __bf16* s, int ld) {
  int lane = threadIdx.x & 31;
  int m = lane & 15, half = lane >> 4;
  const __bf16* r = s + m * ld + half * 8;
  return cat8(*(const v8bf*)(r), *(const v8bf*)(r + 16));
}

__device__ __forceinline__ v16bf load_BfragT(const __bf16* s, int ld) {
  int lane = threadIdx.x & 31;
  int n = lane & 15, kh = (lane >> 4) * 8;
  const __bf16* r = s + n * ld + kh;
  return cat8(*(const v8bf*)(r), *(const v8bf*)(r + 16));
}

__device__ __forceinline__ v8f load_CfragT(const float* s, int ld) {
  int lane = threadIdx.x & 31;
  int n = lane & 15, mh = (lane >> 4) * 8;
  const float* r = s + n * ld + mh;
  v4f lo = *(const v4f*)(r);
  v4f hi = *(const v4f*)(r + 4);
  return __builtin_shufflevector(lo, hi, 0,1,2,3,4,5,6,7);
}

__device__ __forceinline__ void store_CfragT(float* s, int ld, v8f c) {
  int lane = threadIdx.x & 31;
  int n = lane & 15, mh = (lane >> 4) * 8;
  float* r = s + n * ld + mh;
  *(v4f*)(r)     = __builtin_shufflevector(c, c, 0, 1, 2, 3);
  *(v4f*)(r + 4) = __builtin_shufflevector(c, c, 4, 5, 6, 7);
}

__device__ __forceinline__ void store_CfragT_bf(__bf16* s, int ld, v8f c) {
  int lane = threadIdx.x & 31;
  int n = lane & 15, mh = (lane >> 4) * 8;
  v8bf b;
#pragma unroll
  for (int i = 0; i < 8; ++i) b[i] = (__bf16)c[i];
  *(v8bf*)(s + n * ld + mh) = b;
}

#define WMMA_BF16(A, B, C) \
  __builtin_amdgcn_wmma_f32_16x16x32_bf16(false, (A), false, (B), (short)0, (C), false, false)

#if HAVE_TDM
// ---------------- Tensor Data Mover: async 2D bf16 tile -> LDS ------------------------
// D# per CDNA5 ISA.  Rows of `tile0` 2B elements, LDS rows
// padded to 72 elements: pad_interval=32 DWORDs (enc 4), pad_amount=4 DWORDs (enc 3).
// Tile always in-bounds -> tensor dims == tile dims; stride0 in elements.
__device__ __forceinline__ void tdm_load_2d_bf16(
    const __bf16* gptr, const __bf16* ldsptr,
    unsigned tile0, unsigned tile1, unsigned stride0) {
  unsigned long long ga = (unsigned long long)(uintptr_t)gptr;
  unsigned lds = (unsigned)(uintptr_t)ldsptr;          // flat low 32 bits == LDS offset
  u32x4 g0;
  g0[0] = 1u;                                          // count=1, user descriptor
  g0[1] = lds;                                         // lds_addr [63:32]
  g0[2] = (unsigned)ga;                                // global_addr[31:0]
  g0[3] = (unsigned)((ga >> 32) & 0x1FFFFFFu) | (2u << 30);  // addr[56:32] | type=2
  i32x8 g1;
  g1[0] = (1 << 16) | (1 << 20) | (4 << 22) | (3 << 25); // data_size=2B, pad_en, 32dw/4dw
  g1[1] = (int)((tile0 & 0xFFFFu) << 16);                // tensor_dim0[15:0]
  g1[2] = (int)(((tile0 >> 16) & 0xFFFFu) | ((tile1 & 0xFFFFu) << 16)); // dim0 hi | dim1 lo
  g1[3] = (int)(((tile1 >> 16) & 0xFFFFu) | (tile0 << 16));  // dim1 hi | tile_dim0
  g1[4] = (int)tile1;                                    // tile_dim1 | tile_dim2=0
  g1[5] = (int)stride0;                                  // tensor_dim0_stride[31:0]
  g1[6] = 0;
  g1[7] = 0;                                             // dim1_stride unused (2D tile)
  i32x4 z4 = {0, 0, 0, 0};
#if defined(__clang_major__) && __clang_major__ >= 23
  i32x8 z8 = {0, 0, 0, 0, 0, 0, 0, 0};
  __builtin_amdgcn_tensor_load_to_lds(g0, g1, z4, z4, z8, 0);   // 6-arg (clang-23+)
#else
  __builtin_amdgcn_tensor_load_to_lds(g0, g1, z4, z4, 0);       // 5-arg (ROCm 7.2)
#endif
}
#endif

// ---------------- conversion kernel ---------------------------------------------------
__global__ void convert_f32_bf16(const float* __restrict__ X, __bf16* __restrict__ Xb, int n) {
  int i4 = blockIdx.x * 256 + threadIdx.x;
  if (i4 * 4 < n) {
    v4f x = *(const v4f*)(X + (size_t)i4 * 4);
    v4bf o;
#pragma unroll
    for (int j = 0; j < 4; ++j) o[j] = (__bf16)x[j];
    *(v4bf*)(Xb + (size_t)i4 * 4) = o;
  }
}

// ---------------- bf16 WMMA GEMM: C(MxN) = A(MxK) * W(NxK)^T -------------------------
// BM=BN=128, BK=64; 8 waves x (32x64); double-buffered LDS; TDM async tile loads.
__global__ __launch_bounds__(256) void gemm_bf16(
    const __bf16* __restrict__ A, const __bf16* __restrict__ W,
    float* __restrict__ outF, __bf16* __restrict__ outB, __bf16* __restrict__ outB2,
    int M, int N, int Kd, int act) {
  __shared__ __bf16 As [2][128][72];
  __shared__ __bf16 BsT[2][128][72];

  int tid  = threadIdx.x;
  int lane = tid & 31;
  int w    = tid >> 5;
  int wr   = w & 3;
  int wc   = w >> 2;
  int m0   = blockIdx.y * 128;
  int n0   = blockIdx.x * 128;

  const __bf16* Abase = A + (size_t)m0 * Kd;
  const __bf16* Wbase = W + (size_t)n0 * Kd;

  v8f acc[2][4];
#pragma unroll
  for (int i = 0; i < 2; ++i)
#pragma unroll
    for (int j = 0; j < 4; ++j) acc[i][j] = (v8f){};

  auto compute = [&](int buf) {
#pragma unroll
    for (int ks = 0; ks < 2; ++ks) {
      v16bf a0 = load_Afrag(&As[buf][wr * 32][ks * 32], 72);
      v16bf a1 = load_Afrag(&As[buf][wr * 32 + 16][ks * 32], 72);
      v16bf b0 = load_BfragT(&BsT[buf][wc * 64][ks * 32], 72);
      v16bf b1 = load_BfragT(&BsT[buf][wc * 64 + 16][ks * 32], 72);
      v16bf b2 = load_BfragT(&BsT[buf][wc * 64 + 32][ks * 32], 72);
      v16bf b3 = load_BfragT(&BsT[buf][wc * 64 + 48][ks * 32], 72);
      acc[0][0] = WMMA_BF16(a0, b0, acc[0][0]);
      acc[0][1] = WMMA_BF16(a0, b1, acc[0][1]);
      acc[0][2] = WMMA_BF16(a0, b2, acc[0][2]);
      acc[0][3] = WMMA_BF16(a0, b3, acc[0][3]);
      acc[1][0] = WMMA_BF16(a1, b0, acc[1][0]);
      acc[1][1] = WMMA_BF16(a1, b1, acc[1][1]);
      acc[1][2] = WMMA_BF16(a1, b2, acc[1][2]);
      acc[1][3] = WMMA_BF16(a1, b3, acc[1][3]);
    }
  };

#if HAVE_TDM
  // one DMA per tile, issued by a single wave; overlaps the 16 WMMAs of current buffer
  if (w == 0) tdm_load_2d_bf16(Abase, &As [0][0][0], 64, 128, (unsigned)Kd);
  if (w == 1) tdm_load_2d_bf16(Wbase, &BsT[0][0][0], 64, 128, (unsigned)Kd);
  __builtin_amdgcn_s_wait_tensorcnt(0);
  __syncthreads();
  int buf = 0;
  for (int k0 = 0; k0 < Kd; k0 += 64) {
    int nxt = buf ^ 1;
    if (k0 + 64 < Kd) {
      if (w == 0) tdm_load_2d_bf16(Abase + k0 + 64, &As [nxt][0][0], 64, 128, (unsigned)Kd);
      if (w == 1) tdm_load_2d_bf16(Wbase + k0 + 64, &BsT[nxt][0][0], 64, 128, (unsigned)Kd);
    }
    compute(buf);
    __builtin_amdgcn_s_wait_tensorcnt(0);
    __syncthreads();
    buf = nxt;
  }
#else
  // fallback: manual double buffering; global loads of next tile overlap compute
  v8bf ar[4], br[4];
  auto gload = [&](int k0) {
#pragma unroll
    for (int i = 0; i < 4; ++i) {
      int e8 = i * 256 + tid;
      int r = e8 >> 3, c0 = (e8 & 7) * 8;
      ar[i] = *(const v8bf*)(Abase + r * Kd + k0 + c0);
      br[i] = *(const v8bf*)(Wbase + r * Kd + k0 + c0);
    }
  };
  auto lstore = [&](int bufi) {
#pragma unroll
    for (int i = 0; i < 4; ++i) {
      int e8 = i * 256 + tid;
      int r = e8 >> 3, c0 = (e8 & 7) * 8;
      *(v8bf*)(&As [bufi][r][c0]) = ar[i];
      *(v8bf*)(&BsT[bufi][r][c0]) = br[i];
    }
  };
  gload(0); lstore(0);
  __syncthreads();
  int buf = 0;
  for (int k0 = 0; k0 < Kd; k0 += 64) {
    int nxt = buf ^ 1;
    if (k0 + 64 < Kd) gload(k0 + 64);
    compute(buf);
    if (k0 + 64 < Kd) lstore(nxt);
    __syncthreads();
    buf = nxt;
  }
#endif

  int n = lane & 15, mh = (lane >> 4) * 8;
#pragma unroll
  for (int mi = 0; mi < 2; ++mi) {
#pragma unroll
    for (int j = 0; j < 4; ++j) {
#pragma unroll
      for (int i = 0; i < 8; ++i) {
        int gm = m0 + wr * 32 + mi * 16 + mh + i;
        int gn = n0 + wc * 64 + j * 16 + n;
        float v = acc[mi][j][i];
        if (act) v = (v > 0.0f) ? (v + 1.0f) : __expf(v);   // elu(v)+1
        if (outF) outF[(size_t)gm * N + gn] = v;
        int b = gm >> 11, l = gm & 2047;
        int h = gn >> 6,  d = gn & 63;
        if (outB)  outB [(((size_t)(b * NH + h)) * SEQ + l) * HD + d] = (__bf16)v;
        if (outB2) outB2[(((size_t)(b * NH + h)) * HD + d) * SEQ + l] = (__bf16)v;
      }
    }
  }
}

// ---------------- chunked causal linear attention (per b,h block, 8 waves) -----------
__global__ __launch_bounds__(256) void linattn(
    const __bf16* __restrict__ Qg,  const __bf16* __restrict__ Kg,
    const __bf16* __restrict__ KgT, const __bf16* __restrict__ VgT,
    __bf16* __restrict__ Og) {
  __shared__ __bf16 Qs [64][72];
  __shared__ __bf16 Ks [64][72];
  __shared__ __bf16 KsT[64][72];
  __shared__ __bf16 VsT[64][72];
  __shared__ __bf16 Abf[64][72];
  __shared__ __bf16 Sbf[64][72];
  __shared__ float  ST32[64][68];
  __shared__ float  OsT [64][68];
  __shared__ float  ksum[64];
  __shared__ float  norm[64];

  const int tid  = threadIdx.x;
  const int lane = tid & 31;
  const int w    = tid >> 5;
  const int tm   = w >> 1;
  const int tn0  = (w & 1) * 2;
  const int bh   = blockIdx.x;
  const int b    = bh >> 4, h = bh & 15;
  const size_t inbase = (size_t)bh * SEQ * HD;

  for (int e = tid; e < 64 * 68; e += 256) ((float*)ST32)[e] = 0.0f;
  for (int e = tid; e < 64 * 72 / 8; e += 256) ((v8bf*)Sbf)[e] = (v8bf){};
  if (tid < 64) ksum[tid] = 0.0f;
  __syncthreads();

  for (int c = 0; c < SEQ / 64; ++c) {
    // ---- load chunk ----
#if HAVE_TDM
    if (w == 0) tdm_load_2d_bf16(Qg  + inbase + (size_t)c * 64 * HD, &Qs [0][0], 64, 64, HD);
    if (w == 1) tdm_load_2d_bf16(Kg  + inbase + (size_t)c * 64 * HD, &Ks [0][0], 64, 64, HD);
    if (w == 2) tdm_load_2d_bf16(KgT + inbase + (size_t)c * 64,      &KsT[0][0], 64, 64, SEQ);
    if (w == 3) tdm_load_2d_bf16(VgT + inbase + (size_t)c * 64,      &VsT[0][0], 64, 64, SEQ);
    __builtin_amdgcn_s_wait_tensorcnt(0);
#else
    for (int e8 = tid; e8 < 512; e8 += 256) {
      int r = e8 >> 3, d0 = (e8 & 7) * 8;
      size_t g  = inbase + (size_t)(c * 64 + r) * HD + d0;
      size_t gt = inbase + (size_t)r * SEQ + c * 64 + d0;
      v8bf q = *(const v8bf*)(Qg + g), k = *(const v8bf*)(Kg + g);
      v8bf kt = *(const v8bf*)(KgT + gt), vt = *(const v8bf*)(VgT + gt);
      *(v8bf*)(&Qs [r][d0]) = q;  *(v8bf*)(&Ks [r][d0]) = k;
      *(v8bf*)(&KsT[r][d0]) = kt; *(v8bf*)(&VsT[r][d0]) = vt;
    }
#endif
    __syncthreads();

    // ---- phase 1: A = QK^T (masked -> Abf); O_inter = Q*S -> OsT ----
    {
      v16bf a0 = load_Afrag(&Qs[tm * 16][0], 72);
      v16bf a1 = load_Afrag(&Qs[tm * 16][32], 72);
#pragma unroll
      for (int t = 0; t < 2; ++t) {
        int tn = tn0 + t;
        v8f acc = {};
        acc = WMMA_BF16(a0, load_BfragT(&Ks[tn * 16][0], 72), acc);
        acc = WMMA_BF16(a1, load_BfragT(&Ks[tn * 16][32], 72), acc);
        int n = lane & 15, mh = (lane >> 4) * 8;
#pragma unroll
        for (int i = 0; i < 8; ++i) {
          int gi = tm * 16 + mh + i, gj = tn * 16 + n;
          Abf[gi][gj] = (__bf16)((gj <= gi) ? acc[i] : 0.0f);  // causal, keep diag
        }
        v8f oin = {};
        oin = WMMA_BF16(a0, load_BfragT(&Sbf[tn * 16][0], 72), oin);
        oin = WMMA_BF16(a1, load_BfragT(&Sbf[tn * 16][32], 72), oin);
        store_CfragT(&OsT[tn * 16][tm * 16], 68, oin);
      }
    }
    __syncthreads();

    // ---- phase 2: O += A*V ; S += K^T*V (+ bf16 mirror) ; norm pass ----
    {
      v16bf a0 = load_Afrag(&Abf[tm * 16][0], 72);
      v16bf a1 = load_Afrag(&Abf[tm * 16][32], 72);
#pragma unroll
      for (int t = 0; t < 2; ++t) {
        int tn = tn0 + t;
        v8f acc = load_CfragT(&OsT[tn * 16][tm * 16], 68);
        acc = WMMA_BF16(a0, load_BfragT(&VsT[tn * 16][0], 72), acc);
        acc = WMMA_BF16(a1, load_BfragT(&VsT[tn * 16][32], 72), acc);
        store_CfragT(&OsT[tn * 16][tm * 16], 68, acc);
      }
      v16bf ka0 = load_Afrag(&KsT[tm * 16][0], 72);
      v16bf ka1 = load_Afrag(&KsT[tm * 16][32], 72);
#pragma unroll
      for (int t = 0; t < 2; ++t) {
        int tn = tn0 + t;
        v8f sc = load_CfragT(&ST32[tn * 16][tm * 16], 68);
        sc = WMMA_BF16(ka0, load_BfragT(&VsT[tn * 16][0], 72), sc);
        sc = WMMA_BF16(ka1, load_BfragT(&VsT[tn * 16][32], 72), sc);
        store_CfragT(&ST32[tn * 16][tm * 16], 68, sc);
        store_CfragT_bf(&Sbf[tn * 16][tm * 16], 72, sc);
      }
      if (tid < 64) {
        float s = 0.0f;
        for (int j = 0; j < 64; ++j) s += (float)Abf[tid][j];
        float qk = 0.0f;
        for (int d = 0; d < 64; ++d) qk += (float)Qs[tid][d] * ksum[d];
        norm[tid] = s + qk;
      }
    }
    __syncthreads();

    // ---- phase 3: ksum += colsum(K) ; normalized output ----
    if (tid < 64) {
      float s = ksum[tid];
      for (int r = 0; r < 64; ++r) s += (float)KsT[tid][r];
      ksum[tid] = s;
    }
    for (int e8 = tid; e8 < 512; e8 += 256) {
      int r = e8 >> 3, d0 = (e8 & 7) * 8;
      float inv = 1.0f / fmaxf(norm[r], 1e-6f);
      v8bf o;
#pragma unroll
      for (int j = 0; j < 8; ++j) o[j] = (__bf16)(OsT[d0 + j][r] * inv);
      *(v8bf*)(Og + ((size_t)(b * SEQ + c * 64 + r)) * HID + h * HD + d0) = o;
    }
    __syncthreads();
  }
}

// ---------------- launch --------------------------------------------------------------
extern "C" void kernel_launch(void* const* d_in, const int* in_sizes, int n_in,
                              void* d_out, int out_size, void* d_ws, size_t ws_size,
                              hipStream_t stream) {
  const float* x  = (const float*)d_in[0];
  const float* Wq = (const float*)d_in[1];
  const float* Wk = (const float*)d_in[2];
  const float* Wv = (const float*)d_in[3];
  const float* Wo = (const float*)d_in[4];
  float* out = (float*)d_out;

  char* p = (char*)d_ws;
  const size_t WSZ = (size_t)HID * HID * sizeof(__bf16);      // 2 MiB
  const size_t XSZ = (size_t)MTOT * HID * sizeof(__bf16);     // 8 MiB
  __bf16* Wq_b = (__bf16*)(p + 0 * WSZ);
  __bf16* Wk_b = (__bf16*)(p + 1 * WSZ);
  __bf16* Wv_b = (__bf16*)(p + 2 * WSZ);
  __bf16* Wo_b = (__bf16*)(p + 3 * WSZ);
  __bf16* x_bf = (__bf16*)(p + 4 * WSZ);
  __bf16* Qh   = (__bf16*)(p + 4 * WSZ + 1 * XSZ);  // (b,h,l,d)
  __bf16* Kh   = (__bf16*)(p + 4 * WSZ + 2 * XSZ);  // (b,h,l,d)
  __bf16* KhT  = (__bf16*)(p + 4 * WSZ + 3 * XSZ);  // (b,h,d,l)
  __bf16* VhT  = (__bf16*)(p + 4 * WSZ + 4 * XSZ);  // (b,h,d,l)
  __bf16* Attn = (__bf16*)(p + 4 * WSZ + 5 * XSZ);  // (b,l,hid)

  const int WGRID = (HID * HID / 4 + 255) / 256;
  convert_f32_bf16<<<WGRID, 256, 0, stream>>>(Wq, Wq_b, HID * HID);
  convert_f32_bf16<<<WGRID, 256, 0, stream>>>(Wk, Wk_b, HID * HID);
  convert_f32_bf16<<<WGRID, 256, 0, stream>>>(Wv, Wv_b, HID * HID);
  convert_f32_bf16<<<WGRID, 256, 0, stream>>>(Wo, Wo_b, HID * HID);
  convert_f32_bf16<<<(MTOT * HID / 4 + 255) / 256, 256, 0, stream>>>(x, x_bf, MTOT * HID);

  dim3 grid(HID / 128, MTOT / 128);   // (8, 32)
  gemm_bf16<<<grid, 256, 0, stream>>>(x_bf, Wq_b, nullptr, Qh, nullptr, MTOT, HID, HID, 1);
  gemm_bf16<<<grid, 256, 0, stream>>>(x_bf, Wk_b, nullptr, Kh, KhT,     MTOT, HID, HID, 1);
  gemm_bf16<<<grid, 256, 0, stream>>>(x_bf, Wv_b, nullptr, nullptr, VhT, MTOT, HID, HID, 0);

  linattn<<<BATCH * NH, 256, 0, stream>>>(Qh, Kh, KhT, VhT, Attn);

  gemm_bf16<<<grid, 256, 0, stream>>>(Attn, Wo_b, out, nullptr, nullptr, MTOT, HID, HID, 0);
}
